// MLHAAttention_27908697489569
// MI455X (gfx1250) — matrix operand order repaired and checked by reading, not compiled, not run on hardware
//
#include <hip/hip_runtime.h>
#include <hip/hip_bf16.h>

typedef __bf16 bf16_t;
typedef __attribute__((ext_vector_type(16))) __bf16 v16bf;
typedef __attribute__((ext_vector_type(8)))  __bf16 v8bf;
typedef __attribute__((ext_vector_type(8)))  float  v8f;
typedef __attribute__((ext_vector_type(4)))  int    v4i;

#define HIDDEN 1024
#define NHEADS 16
#define HDIM   64
#define BATCH  2
#define SEQ    2048
#define TOKENS (BATCH * SEQ)   // 4096
#define NEGMIN (-3.402823466e38f)

// ---------------- async global->LDS copy (CDNA5), with safe fallback --------

#if __has_builtin(__builtin_amdgcn_global_load_async_to_lds_b128)
#define HAS_ASYNC_LDS 1
typedef __attribute__((address_space(1))) v4i* as1_v4i;
typedef __attribute__((address_space(3))) v4i* as3_v4i;
#endif

// copy 16 bytes (8 bf16) per lane, global -> LDS
__device__ inline void cp16(const bf16_t* g, bf16_t* l) {
#ifdef HAS_ASYNC_LDS
  __builtin_amdgcn_global_load_async_to_lds_b128(
      (as1_v4i)(uintptr_t)g, (as3_v4i)(uint32_t)(uintptr_t)l, 0, 0);
#else
  *(v8bf*)l = *(const v8bf*)g;
#endif
}

__device__ inline void wait_async_0() {
#ifdef HAS_ASYNC_LDS
#if __has_builtin(__builtin_amdgcn_s_wait_asynccnt)
  __builtin_amdgcn_s_wait_asynccnt(0);
#else
  asm volatile("s_wait_asynccnt 0x0" ::: "memory");
#endif
#endif
}
__device__ inline void wait_async_1() {
#ifdef HAS_ASYNC_LDS
#if __has_builtin(__builtin_amdgcn_s_wait_asynccnt)
  __builtin_amdgcn_s_wait_asynccnt(1);
#else
  asm volatile("s_wait_asynccnt 0x1" ::: "memory");
#endif
#endif
}
__device__ inline void wait_async_2() {
#ifdef HAS_ASYNC_LDS
#if __has_builtin(__builtin_amdgcn_s_wait_asynccnt)
  __builtin_amdgcn_s_wait_asynccnt(2);
#else
  asm volatile("s_wait_asynccnt 0x2" ::: "memory");
#endif
#endif
}

// ---------------- fragment loaders (CDNA5 WMMA 16x16x32 bf16 layouts) -------

__device__ inline v8bf load8(const bf16_t* p) { return *(const v8bf*)p; }

__device__ inline v16bf cat16(v8bf lo, v8bf hi) {
  return __builtin_shufflevector(lo, hi, 0,1,2,3,4,5,6,7,8,9,10,11,12,13,14,15);
}

// A operand (16xK=32): lane&15 = row M; half hi picks K {hi*8..+7, hi*8+16..+23}
__device__ inline v16bf load_fragA(const bf16_t* base, int ld, int lane) {
  int row = lane & 15, hi = lane >> 4;
  const bf16_t* p = base + row * ld + hi * 8;
  return cat16(load8(p), load8(p + 16));
}

// B operand (K=32 x 16) given B^T row-major [N][K]: lane&15 = col N; half hi -> K {hi*8..+7, hi*8+16..+23}
__device__ inline v16bf load_fragB(const bf16_t* baseT, int ld, int lane) {
  int row = lane & 15, hi = lane >> 4;
  const bf16_t* p = baseT + row * ld + hi * 8;
  return cat16(load8(p), load8(p + 16));
}

__device__ inline v8f wmma_bf16(v16bf a, v16bf b, v8f c) {
  return __builtin_amdgcn_wmma_f32_16x16x32_bf16(false, a, false, b, (short)0, c, false, false);
}

// ---------------- conversion / transpose ------------------------------------

__global__ void k_f32_to_bf16(const float* __restrict__ in, bf16_t* __restrict__ out, int n) {
  int i = blockIdx.x * 256 + threadIdx.x;
  if (i < n) out[i] = (bf16_t)in[i];
}

// V [TOKENS, 64] -> Vt [B][64][SEQ]
__global__ void k_transpose_v(const bf16_t* __restrict__ V, bf16_t* __restrict__ Vt) {
  int i = blockIdx.x * 256 + threadIdx.x;
  if (i >= TOKENS * HDIM) return;
  int d = i & 63, tok = i >> 6, b = tok >> 11, s = tok & 2047;
  Vt[(b * HDIM + d) * SEQ + s] = V[i];
}

// ---------------- GEMM: C[M,N] = alpha * A[M,K] @ W[N,K]^T (bf16 in, f32 acc) ----
// block = 256 threads = 8 waves; tile 128(M) x 64(N);
// W panel double-buffered in LDS via async global->LDS copies.

template <bool F32OUT>
__global__ __launch_bounds__(256) void k_gemm_xwt(
    const bf16_t* __restrict__ A, const bf16_t* __restrict__ W, void* __restrict__ Cout,
    int M, int N, int K, float alpha) {
  __shared__ __align__(16) bf16_t Wl[2][64 * 40];   // padded stride 40 (16B aligned)
  const int lane = threadIdx.x & 31;
  const int wv   = threadIdx.x >> 5;
  const int mbase = blockIdx.x * 128 + wv * 16;
  const int nbase = blockIdx.y * 64;
  const int hi = lane >> 4, n0 = lane & 15;

  v8f acc[4];
  for (int j = 0; j < 4; ++j) acc[j] = (v8f){0,0,0,0,0,0,0,0};

  const int nl = threadIdx.x >> 2;        // 0..63
  const int kg = (threadIdx.x & 3) * 8;   // 0,8,16,24

  const int iters = K >> 5;
  // prologue: stage panel 0
  cp16(W + (size_t)(nbase + nl) * K + kg, &Wl[0][nl * 40 + kg]);

  for (int it = 0; it < iters; ++it) {
    const int cur = it & 1;
    if (it + 1 < iters) {
      cp16(W + (size_t)(nbase + nl) * K + (it + 1) * 32 + kg, &Wl[cur ^ 1][nl * 40 + kg]);
      wait_async_1();   // async ops retire in order -> panel `cur` has landed
    } else {
      wait_async_0();
    }
    __syncthreads();

    v16bf af = load_fragA(A + (size_t)mbase * K + it * 32, K, lane);
    for (int jn = 0; jn < 4; ++jn) {
      v16bf bf = load_fragB(&Wl[cur][jn * 16 * 40], 40, lane);
      acc[jn] = wmma_bf16(af, bf, acc[jn]);
    }
    __syncthreads();    // protect buffer being refilled next iteration
  }

  for (int jn = 0; jn < 4; ++jn) {
    int n = nbase + jn * 16 + n0;
    for (int r = 0; r < 8; ++r) {
      int m = mbase + r + 8 * hi;
      float v = acc[jn][r] * alpha;
      if (F32OUT) ((float*)Cout)[(size_t)m * N + n] = v;
      else        ((bf16_t*)Cout)[(size_t)m * N + n] = (bf16_t)v;
    }
  }
}

// ---------------- flash attention --------------------------------------------
// One wave per 16-query tile; 8 waves/block share one batch's K/V chunks via
// double-buffered async LDS staging (MQA: K/V shared across heads -> 8x reuse).
// S^T = K_chunk(16k x 64d) @ Q^T(64d x 16q) so P^T feeds the B operand of
// O^T = V^T @ P^T with only a shfl_xor(16) half-wave exchange.

__global__ __launch_bounds__(256) void k_attention(
    const bf16_t* __restrict__ Q,   // [TOKENS, 1024] (h*64+d), pre-scaled by D^-0.5
    const bf16_t* __restrict__ Kb,  // [TOKENS, 64]
    const bf16_t* __restrict__ Vt,  // [B][64][SEQ]
    const float*  __restrict__ mask,// [B, SEQ]
    bf16_t* __restrict__ Out)       // [TOKENS, 1024]
{
  __shared__ __align__(16) bf16_t Kl[2][32 * 72];   // 32 keys x 64 d, stride 72
  __shared__ __align__(16) bf16_t Vl[2][64 * 40];   // 64 d x 32 keys, stride 40

  const int tid  = threadIdx.x;
  const int lane = tid & 31;
  const int wid  = blockIdx.x * 8 + (tid >> 5);           // 4096 tasks
  const int qt = wid & 127, h = (wid >> 7) & 15, b = wid >> 11;  // block shares b
  const int hi = lane >> 4, n0 = lane & 15;
  const int qrow = qt * 16 + n0;

  // Q fragments: B operand, d-chunks 0 and 1
  const bf16_t* qp = Q + (size_t)(b * SEQ + qrow) * HIDDEN + h * HDIM + hi * 8;
  const v16bf qf0 = cat16(load8(qp),      load8(qp + 16));
  const v16bf qf1 = cat16(load8(qp + 32), load8(qp + 48));

  const bf16_t* kbase = Kb + (size_t)b * SEQ * HDIM;
  const bf16_t* vtb   = Vt + (size_t)b * HDIM * SEQ;
  const float*  mrow  = mask + b * SEQ;

  // staging assignments: 256 threads x 16B cover each 4KB chunk
  const int krow = tid >> 3, kseg = (tid & 7) * 8;   // K: 32 rows x 64
  const int vrow = tid >> 2, vseg = (tid & 3) * 8;   // V: 64 rows x 32

  float m_run = -INFINITY, l_run = 0.f;
  v8f acc[4];
  for (int j = 0; j < 4; ++j) acc[j] = (v8f){0,0,0,0,0,0,0,0};

  // prologue: stage chunk 0
  cp16(kbase + (size_t)krow * HDIM + kseg, &Kl[0][krow * 72 + kseg]);
  cp16(vtb + (size_t)vrow * SEQ + vseg,    &Vl[0][vrow * 40 + vseg]);

  for (int it = 0; it < SEQ / 32; ++it) {
    const int kc = it * 32, cur = it & 1;
    if (it + 1 < SEQ / 32) {
      cp16(kbase + (size_t)(kc + 32 + krow) * HDIM + kseg, &Kl[cur ^ 1][krow * 72 + kseg]);
      cp16(vtb + (size_t)vrow * SEQ + kc + 32 + vseg,      &Vl[cur ^ 1][vrow * 40 + vseg]);
      wait_async_2();   // the 2 older copies (chunk `cur`) have landed
    } else {
      wait_async_0();
    }
    __syncthreads();

    const bf16_t* kl = Kl[cur];
    const bf16_t* vl = Vl[cur];

    // --- scores S^T: two 16(key)x16(q) tiles, K-dim = d (2 chunks of 32)
    const bf16_t* k0p = kl + n0 * 72 + hi * 8;        // keys kc..kc+15
    const bf16_t* k1p = k0p + 16 * 72;                 // keys kc+16..kc+31
    v16bf ka0 = cat16(load8(k0p),      load8(k0p + 16));
    v16bf ka1 = cat16(load8(k0p + 32), load8(k0p + 48));
    v16bf kb0 = cat16(load8(k1p),      load8(k1p + 16));
    v16bf kb1 = cat16(load8(k1p + 32), load8(k1p + 48));

    v8f z = (v8f){0,0,0,0,0,0,0,0};
    v8f st0 = wmma_bf16(ka0, qf0, z);  st0 = wmma_bf16(ka1, qf1, st0);
    v8f st1 = wmma_bf16(kb0, qf0, z);  st1 = wmma_bf16(kb1, qf1, st1);

    // --- mask bias: row r of tile = key kc + (tile?16:0) + hi*8 + r
    float4 m0a = *(const float4*)(mrow + kc + hi * 8);
    float4 m0b = *(const float4*)(mrow + kc + hi * 8 + 4);
    float4 m1a = *(const float4*)(mrow + kc + 16 + hi * 8);
    float4 m1b = *(const float4*)(mrow + kc + 16 + hi * 8 + 4);
    float s0[8], s1[8];
    s0[0]=st0[0]+(1.f-m0a.x)*NEGMIN; s0[1]=st0[1]+(1.f-m0a.y)*NEGMIN;
    s0[2]=st0[2]+(1.f-m0a.z)*NEGMIN; s0[3]=st0[3]+(1.f-m0a.w)*NEGMIN;
    s0[4]=st0[4]+(1.f-m0b.x)*NEGMIN; s0[5]=st0[5]+(1.f-m0b.y)*NEGMIN;
    s0[6]=st0[6]+(1.f-m0b.z)*NEGMIN; s0[7]=st0[7]+(1.f-m0b.w)*NEGMIN;
    s1[0]=st1[0]+(1.f-m1a.x)*NEGMIN; s1[1]=st1[1]+(1.f-m1a.y)*NEGMIN;
    s1[2]=st1[2]+(1.f-m1a.z)*NEGMIN; s1[3]=st1[3]+(1.f-m1a.w)*NEGMIN;
    s1[4]=st1[4]+(1.f-m1b.x)*NEGMIN; s1[5]=st1[5]+(1.f-m1b.y)*NEGMIN;
    s1[6]=st1[6]+(1.f-m1b.z)*NEGMIN; s1[7]=st1[7]+(1.f-m1b.w)*NEGMIN;

    // --- online softmax over keys (per lane: fixed q column)
    float locmax = -INFINITY;
    for (int r = 0; r < 8; ++r) locmax = fmaxf(locmax, fmaxf(s0[r], s1[r]));
    locmax = fmaxf(locmax, __shfl_xor(locmax, 16, 32));
    float newm  = fmaxf(m_run, locmax);
    float alpha = __expf(m_run - newm);
    float p0[8], p1[8]; float lsum = 0.f;
    for (int r = 0; r < 8; ++r) {
      p0[r] = __expf(s0[r] - newm);
      p1[r] = __expf(s1[r] - newm);
      lsum += p0[r] + p1[r];
    }
    lsum += __shfl_xor(lsum, 16, 32);
    l_run = l_run * alpha + lsum;
    m_run = newm;
    for (int j = 0; j < 4; ++j) acc[j] *= alpha;

    // --- build P^T B-operand (half-wave exchange)
    v16bf pb;
    for (int r = 0; r < 8; ++r) {
      float o0 = __shfl_xor(p0[r], 16, 32);
      float o1 = __shfl_xor(p1[r], 16, 32);
      pb[r]     = (bf16_t)(hi ? o1 : p0[r]);
      pb[r + 8] = (bf16_t)(hi ? p1[r] : o0);
    }

    // --- O^T accumulation: 4 tiles of 16(d) x 16(q), A = V^T chunk (from LDS)
    for (int j = 0; j < 4; ++j) {
      const bf16_t* vp = vl + (16 * j + n0) * 40 + hi * 16;
      v16bf vf = cat16(load8(vp), load8(vp + 8));
      acc[j] = wmma_bf16(vf, pb, acc[j]);
    }
    __syncthreads();    // protect buffer being refilled next iteration
  }

  // --- normalize and store bf16 [token, h*64 + d]
  float inv = 1.f / l_run;
  int tok = b * SEQ + qt * 16 + n0;
  bf16_t* op = Out + (size_t)tok * HIDDEN + h * HDIM + 8 * hi;
  for (int j = 0; j < 4; ++j) {
    v8bf ov;
    for (int r = 0; r < 8; ++r) ov[r] = (bf16_t)(acc[j][r] * inv);
    *(v8bf*)(op + 16 * j) = ov;
  }
}

// ---------------- host orchestration ----------------------------------------

extern "C" void kernel_launch(void* const* d_in, const int* in_sizes, int n_in,
                              void* d_out, int out_size, void* d_ws, size_t ws_size,
                              hipStream_t stream) {
  const float* hs   = (const float*)d_in[0];  // [2,2048,1024]
  const float* mask = (const float*)d_in[1];  // [2,2048]
  const float* wq   = (const float*)d_in[2];  // [1024,1024]
  const float* wk   = (const float*)d_in[3];  // [64,1024]
  const float* wv   = (const float*)d_in[4];  // [64,1024]
  const float* wo   = (const float*)d_in[5];  // [1024,1024]

  char* ws = (char*)d_ws;
  size_t off = 0;
  auto alloc = [&](size_t elems) { bf16_t* p = (bf16_t*)(ws + off); off += ((elems * 2 + 255) & ~size_t(255)); return p; };
  bf16_t* Xbf  = alloc((size_t)TOKENS * HIDDEN);   // 8 MB
  bf16_t* Wq   = alloc((size_t)HIDDEN * HIDDEN);   // 2 MB
  bf16_t* Wk   = alloc((size_t)HDIM * HIDDEN);
  bf16_t* Wv   = alloc((size_t)HDIM * HIDDEN);
  bf16_t* Wo   = alloc((size_t)HIDDEN * HIDDEN);   // 2 MB
  bf16_t* Qbf  = alloc((size_t)TOKENS * HIDDEN);   // 8 MB
  bf16_t* Kbf  = alloc((size_t)TOKENS * HDIM);
  bf16_t* Vbf  = alloc((size_t)TOKENS * HDIM);
  bf16_t* Vt   = alloc((size_t)TOKENS * HDIM);
  bf16_t* Attn = alloc((size_t)TOKENS * HIDDEN);   // 8 MB  (~30 MB total)

  auto cvt = [&](const float* in, bf16_t* out, int n) {
    k_f32_to_bf16<<<(n + 255) / 256, 256, 0, stream>>>(in, out, n);
  };
  cvt(hs, Xbf, TOKENS * HIDDEN);
  cvt(wq, Wq, HIDDEN * HIDDEN);
  cvt(wk, Wk, HDIM * HIDDEN);
  cvt(wv, Wv, HDIM * HIDDEN);
  cvt(wo, Wo, HIDDEN * HIDDEN);

  const float scaling = 0.125f;  // 64^-0.5, folded into Q
  k_gemm_xwt<false><<<dim3(TOKENS / 128, HIDDEN / 64), 256, 0, stream>>>(Xbf, Wq, Qbf, TOKENS, HIDDEN, HIDDEN, scaling);
  k_gemm_xwt<false><<<dim3(TOKENS / 128, 1),           256, 0, stream>>>(Xbf, Wk, Kbf, TOKENS, HDIM,   HIDDEN, 1.0f);
  k_gemm_xwt<false><<<dim3(TOKENS / 128, 1),           256, 0, stream>>>(Xbf, Wv, Vbf, TOKENS, HDIM,   HIDDEN, 1.0f);

  k_transpose_v<<<(TOKENS * HDIM + 255) / 256, 256, 0, stream>>>(Vbf, Vt);

  k_attention<<<(BATCH * NHEADS * (SEQ / 16)) / 8, 256, 0, stream>>>(Qbf, Kbf, Vt, mask, Attn);

  k_gemm_xwt<true><<<dim3(TOKENS / 128, HIDDEN / 64), 256, 0, stream>>>(Attn, Wo, d_out, TOKENS, HIDDEN, HIDDEN, 1.0f);
}
